// SSN_DeepLab_38079180046743
// MI455X (gfx1250) — matrix operand order repaired and checked by reading, not compiled, not run on hardware
//
#include <hip/hip_runtime.h>
#include <hip/hip_bf16.h>
#include <stddef.h>
#include <stdint.h>

// ---------------------------------------------------------------------------
// Problem constants (from the reference)
// ---------------------------------------------------------------------------
#define B_ 2
#define H_ 256
#define W_ 256
#define HW_ 65536            // H_*W_
#define CBB_ 1029
#define N1_ 500
#define N2_ 100
#define N3_ 15               // FEATURE_DIM - 5
#define FDIM_ 20
#define NSPIX_ 256           // 16*16
#define NITER_ 10

// GEMM tiling: block = 256 threads (8 waves), tile M=128 x N=128, K-step 32.
// TN=128 halves im2col A-tile traffic vs TN=64 (L2-roofline driven) and gives
// each wave 8 back-to-back WMMAs per K-step.
#define TM 128
#define TN 128
#define TK 32
#define BSTRIDE 40           // LDS row stride (ushorts) for transposed B panel (80B, 16B-aligned)

// K / N padding for the three convs (K = Cin*9, Kpad multiple of 32, Npad multiple of TN)
#define K1_ 9261
#define KP1_ 9280
#define NP1_ 512
#define K2_ 4500
#define KP2_ 4512
#define NP2_ 128
#define K3_ 900
#define KP3_ 928
#define NP3_ 128

typedef __attribute__((ext_vector_type(16))) __bf16 v16bf;
typedef __attribute__((ext_vector_type(8)))  float  v8f;

// ---------------------------------------------------------------------------
// bf16 <-> f32 helpers (bit-level, round-to-nearest-even)
// ---------------------------------------------------------------------------
__device__ __forceinline__ unsigned short f_to_bfu(float f) {
    union { float f; unsigned u; } x; x.f = f;
    unsigned r = (x.u + 0x7FFFu + ((x.u >> 16) & 1u)) >> 16;
    return (unsigned short)r;
}
__device__ __forceinline__ __bf16 us_to_bf(unsigned short u) {
    union { unsigned short u; __bf16 b; } t; t.u = u; return t.b;
}

// ---------------------------------------------------------------------------
// CDNA5 async global->LDS copy: each active lane moves 16B from its own
// global address to its own LDS byte offset. Tracked by ASYNCcnt.
// ---------------------------------------------------------------------------
__device__ __forceinline__ void async_b128_to_lds(unsigned lds_off, const void* gptr) {
    asm volatile("global_load_async_to_lds_b128 %0, %1, off"
                 :: "v"(lds_off), "v"((unsigned long long)(uintptr_t)gptr)
                 : "memory");
}
__device__ __forceinline__ void wait_asynccnt0() {
    asm volatile("s_wait_asynccnt 0x0" ::: "memory");
}

// ---------------------------------------------------------------------------
// Generic f32 -> bf16(ushort) conversion
// ---------------------------------------------------------------------------
__global__ __launch_bounds__(256) void cvt_bf16_kernel(const float* __restrict__ src,
                                                       unsigned short* __restrict__ dst,
                                                       size_t n) {
    size_t i = (size_t)blockIdx.x * 256 + threadIdx.x;
    if (i < n) dst[i] = f_to_bfu(src[i]);
}

// ---------------------------------------------------------------------------
// Repack conv weights W[o][c][3][3] (f32) -> wnk[n = o][k = c*9+j] (bf16),
// N-major, zero-padded to [Npad][Kpad]. A (n, k0) weight panel is then 128
// rows x 64B, ideal for per-lane async b128 gathers (transposing into LDS).
// ---------------------------------------------------------------------------
__global__ __launch_bounds__(256) void repack_w_kernel(const float* __restrict__ w,
                                                       unsigned short* __restrict__ wnk,
                                                       int Cin, int Nout, int K,
                                                       int Kpad, int Npad) {
    size_t idx = (size_t)blockIdx.x * 256 + threadIdx.x;
    size_t total = (size_t)Kpad * Npad;
    if (idx >= total) return;
    int n  = (int)(idx / Kpad);
    int kk = (int)(idx % Kpad);
    unsigned short v = 0;
    if (kk < K && n < Nout) {
        int c = kk / 9, j = kk - c * 9;
        v = f_to_bfu(w[((size_t)n * Cin + c) * 9 + j]);
    }
    wnk[idx] = v;
}

// ---------------------------------------------------------------------------
// Implicit-GEMM 3x3 SAME conv using v_wmma_f32_16x16x32_bf16.
//   in  : [B][Cin][H][W]  bf16
//   wnk : [Npad][Kpad]    bf16 (N-major packed weights)
//   out : [B][Nout][HW]   f32
// A tile im2col'd to LDS by all threads; B panel (8KB) staged by waves 0-1
// with global_load_async_to_lds_b128 into a transposed, padded LDS layout so
// each lane's fragment is 16 contiguous ushorts (2 x ds_load_b128).
// Each wave: 16 M-rows x 128 N-cols = 8 WMMA accumulators.
// ---------------------------------------------------------------------------
__global__ __launch_bounds__(256) void conv3x3_wmma_kernel(
        const unsigned short* __restrict__ in,
        const unsigned short* __restrict__ wnk,
        float* __restrict__ out,
        int Cin, int K, int Kpad, int Npad, int Nout) {
    __shared__ unsigned short lA[TM * TK];          // 128 x 32 row-major (M x K), 8KB
    __shared__ unsigned short lBT[TN * BSTRIDE];    // 128 rows (N) x 32 K, padded, 10KB

    const int tid  = threadIdx.x;
    const int lane = tid & 31;
    const int wave = tid >> 5;
    const int m0   = blockIdx.x * TM;
    const int n0   = blockIdx.y * TN;
    const int row  = lane & 15;   // M-row within 16 (A), N-col within 16 (B/D)
    const int half = lane >> 4;

    const unsigned lbt_base = (unsigned)(uintptr_t)&lBT[0];

    v8f acc[8];
#pragma unroll
    for (int s = 0; s < 8; ++s)
#pragma unroll
        for (int r = 0; r < 8; ++r) acc[s][r] = 0.0f;

    for (int k0 = 0; k0 < Kpad; k0 += TK) {
        // ---- stage B panel: waves 0-1 issue 16 async b128 gathers (8KB) ----
        if (wave < 2) {
#pragma unroll
            for (int i = 0; i < 8; ++i) {
                int chunk = wave * 256 + i * 32 + lane;   // 0..511
                int n     = chunk >> 2;                   // 0..127
                int part  = chunk & 3;                    // 16B chunk within 64B row
                unsigned lds_off = lbt_base + (unsigned)(n * (BSTRIDE * 2) + part * 16);
                const void* g = wnk + (size_t)(n0 + n) * Kpad + k0 + part * 8;
                async_b128_to_lds(lds_off, g);
            }
        }
        // ---- stage A tile (im2col on the fly, zero pad for SAME) ----
#pragma unroll 4
        for (int i = tid; i < TM * TK; i += 256) {
            int ar = i >> 5;          // 0..127
            int ak = i & 31;
            int kk = k0 + ak;
            unsigned short v = 0;
            if (kk < K) {
                int m  = m0 + ar;
                int b  = m >> 16;     // HW_ == 65536
                int p  = m & 0xFFFF;
                int py = p >> 8;      // W_ == 256
                int px = p & 255;
                int c  = kk / 9;
                int j  = kk - c * 9;
                int dy = j / 3 - 1;
                int dx = j - (j / 3) * 3 - 1;
                int yy = py + dy, xx = px + dx;
                if ((unsigned)yy < (unsigned)H_ && (unsigned)xx < (unsigned)W_)
                    v = in[((size_t)(b * Cin + c) << 16) + (yy << 8) + xx];
            }
            lA[i] = v;
        }
        if (wave < 2) wait_asynccnt0();
        __syncthreads();

        // prefetch next weight panel into cache (global_prefetch_b8)
        if (k0 + TK < Kpad)
            __builtin_prefetch(wnk + (size_t)(n0 + (tid & 127)) * Kpad + k0 + TK, 0, 0);

        // ---- A fragment (16x32 bf16, K = half*16 + e, matches B) ----
        v16bf afrag;
#pragma unroll
        for (int e = 0; e < 16; ++e) {
            int kk = half * 16 + e;
            afrag[e] = us_to_bf(lA[(wave * 16 + row) * TK + kk]);
        }
        // ---- 8 WMMAs: 16x16 output subtiles across N ----
#pragma unroll
        for (int s = 0; s < 8; ++s) {
            // lane's 16 B elements: column (s*16+row), K = half*16 + e -> contiguous
            const unsigned short* bsrc = &lBT[(s * 16 + row) * BSTRIDE + half * 16];
            v16bf bfrag;
#pragma unroll
            for (int e = 0; e < 16; ++e) bfrag[e] = us_to_bf(bsrc[e]);
            acc[s] = __builtin_amdgcn_wmma_f32_16x16x32_bf16(
                         false, afrag, false, bfrag,
                         (short)0, acc[s], false, false);
        }
        __syncthreads();
    }

    // ---- store D (lane L: N = L%16, VGPR r -> M = r + 8*(L/16)) ----
#pragma unroll
    for (int s = 0; s < 8; ++s) {
        int n = n0 + s * 16 + row;
        if (n >= Nout) continue;
#pragma unroll
        for (int r = 0; r < 8; ++r) {
            int m = m0 + wave * 16 + r + half * 8;
            int b = m >> 16, p = m & 0xFFFF;
            out[((size_t)(b * Nout + n) << 16) + p] = acc[s][r];
        }
    }
}

// ---------------------------------------------------------------------------
// BatchNorm statistics: one block per channel, reduce over B*HW
// stats[c] = mean, stats[C+c] = rsqrt(var+eps)
// ---------------------------------------------------------------------------
__global__ __launch_bounds__(256) void bn_stats_kernel(const float* __restrict__ y,
                                                       float* __restrict__ stats, int C) {
    int c = blockIdx.x, t = threadIdx.x;
    float s = 0.f, s2 = 0.f;
    for (int i = t; i < B_ * HW_; i += 256) {
        int b = i >> 16, p = i & 0xFFFF;
        float v = y[((size_t)(b * C + c) << 16) + p];
        s += v; s2 += v * v;
    }
    __shared__ float rs[256], rq[256];
    rs[t] = s; rq[t] = s2; __syncthreads();
    for (int off = 128; off > 0; off >>= 1) {
        if (t < off) { rs[t] += rs[t + off]; rq[t] += rq[t + off]; }
        __syncthreads();
    }
    if (t == 0) {
        float inv = 1.0f / (float)(B_ * HW_);
        float mu  = rs[0] * inv;
        float var = rq[0] * inv - mu * mu;
        stats[c] = mu;
        stats[C + c] = rsqrtf(var + 1e-5f);
    }
}

// BN + ReLU -> bf16 (next conv input)
__global__ __launch_bounds__(256) void bn_relu_bf16_kernel(
        const float* __restrict__ y, const float* __restrict__ stats,
        const float* __restrict__ g, const float* __restrict__ beta,
        unsigned short* __restrict__ o, int C) {
    size_t idx = (size_t)blockIdx.x * 256 + threadIdx.x;
    int c = (int)((idx >> 16) % C);
    float v = (y[idx] - stats[c]) * stats[C + c] * g[c] + beta[c];
    o[idx] = f_to_bfu(fmaxf(v, 0.0f));
}

// BN + ReLU -> f32 into pixel_f channels [0,15)
__global__ __launch_bounds__(256) void bn_relu_pf_kernel(
        const float* __restrict__ y, const float* __restrict__ stats,
        const float* __restrict__ g, const float* __restrict__ beta,
        float* __restrict__ pf) {
    size_t idx = (size_t)blockIdx.x * 256 + threadIdx.x;   // over B*15*HW
    int bc = (int)(idx >> 16);
    int b = bc / N3_, c = bc % N3_;
    int p = (int)(idx & 0xFFFF);
    float v = (y[idx] - stats[c]) * stats[N3_ + c] * g[c] + beta[c];
    pf[((size_t)(b * FDIM_ + c) << 16) + p] = fmaxf(v, 0.0f);
}

// concat x (3ch) + coords (2ch) into pixel_f channels [15,20)
__global__ __launch_bounds__(256) void concat_pf_kernel(
        const float* __restrict__ x, const float* __restrict__ coords,
        float* __restrict__ pf) {
    size_t idx = (size_t)blockIdx.x * 256 + threadIdx.x;   // over B*5*HW
    int bc = (int)(idx >> 16);
    int b = bc / 5, c5 = bc % 5;
    int p = (int)(idx & 0xFFFF);
    float v = (c5 < 3) ? x[((size_t)(b * 3 + c5) << 16) + p]
                       : coords[((size_t)(b * 2 + (c5 - 3)) << 16) + p];
    pf[((size_t)(b * FDIM_ + 15 + c5) << 16) + p] = v;
}

// initial centroids: mean over each 16x16 block
__global__ __launch_bounds__(256) void cent_init_kernel(const float* __restrict__ pf,
                                                        float* __restrict__ cent) {
    int b = blockIdx.x >> 8;
    int s = blockIdx.x & 255;
    int sy = s >> 4, sx = s & 15;
    int t = threadIdx.x;
    int py = (sy << 4) + (t >> 4), px = (sx << 4) + (t & 15);
    int p = (py << 8) + px;
    float pv[FDIM_];
#pragma unroll
    for (int c = 0; c < FDIM_; ++c)
        pv[c] = pf[((size_t)(b * FDIM_ + c) << 16) + p];
    __shared__ float red[256];
#pragma unroll 1
    for (int c = 0; c < FDIM_; ++c) {
        red[t] = pv[c]; __syncthreads();
        for (int off = 128; off > 0; off >>= 1) {
            if (t < off) red[t] += red[t + off];
            __syncthreads();
        }
        if (t == 0) cent[((size_t)(b * FDIM_ + c) << 8) + s] = red[0] * (1.0f / 256.0f);
        __syncthreads();
    }
}

__global__ __launch_bounds__(256) void zero_kernel(float* __restrict__ p, size_t n) {
    size_t i = (size_t)blockIdx.x * 256 + threadIdx.x;
    if (i < n) p[i] = 0.0f;
}

// ---------------------------------------------------------------------------
// One SSN iteration: per 16x16 pixel tile (= 1 superpixel cell), all pixels
// share the same 9 candidate superpixels -> reduce in LDS (ds_add_f32),
// flush 189 global atomics per block.
// ---------------------------------------------------------------------------
__global__ __launch_bounds__(256) void ssn_step_kernel(
        const float* __restrict__ pf, const float* __restrict__ cent,
        float* __restrict__ num, float* __restrict__ den,
        float* __restrict__ aff_out) {
    int b = blockIdx.x >> 8;
    int s = blockIdx.x & 255;
    int sy = s >> 4, sx = s & 15;
    int t = threadIdx.x;
    int py = (sy << 4) + (t >> 4), px = (sx << 4) + (t & 15);
    int p = (py << 8) + px;

    float pv[FDIM_];
#pragma unroll
    for (int c = 0; c < FDIM_; ++c)
        pv[c] = pf[((size_t)(b * FDIM_ + c) << 16) + p];

    __shared__ float centl[9][FDIM_];
    __shared__ float numl[9][FDIM_];
    __shared__ float denl[9];

    if (t < 9 * FDIM_) {
        int j = t / FDIM_, c = t % FDIM_;
        int dr = j / 3 - 1, dc = j % 3 - 1;
        int rr = sy + dr, cc = sx + dc;
        bool val = ((unsigned)rr < 16u) && ((unsigned)cc < 16u);
        int rrc = rr < 0 ? 0 : (rr > 15 ? 15 : rr);
        int ccc = cc < 0 ? 0 : (cc > 15 ? 15 : cc);
        int cid = rrc * 16 + ccc;
        centl[j][c] = val ? cent[((size_t)(b * FDIM_ + c) << 8) + cid] : 0.0f;
        numl[j][c] = 0.0f;
    }
    if (t < 9) denl[t] = 0.0f;
    __syncthreads();

    float logit[9];
    bool  vld[9];
#pragma unroll
    for (int j = 0; j < 9; ++j) {
        int dr = j / 3 - 1, dc = j % 3 - 1;
        int rr = sy + dr, cc = sx + dc;
        vld[j] = ((unsigned)rr < 16u) && ((unsigned)cc < 16u);
        if (vld[j]) {
            float d = 0.0f;
#pragma unroll
            for (int c = 0; c < FDIM_; ++c) {
                float df = pv[c] - centl[j][c];
                d += df * df;
            }
            logit[j] = -d;
        } else {
            logit[j] = -3.0e38f;
        }
    }
    float mx = logit[0];
#pragma unroll
    for (int j = 1; j < 9; ++j) mx = fmaxf(mx, logit[j]);
    float ex[9], sum = 0.0f;
#pragma unroll
    for (int j = 0; j < 9; ++j) { ex[j] = expf(logit[j] - mx); sum += ex[j]; }
    float inv = 1.0f / sum;
#pragma unroll
    for (int j = 0; j < 9; ++j) {
        float a = ex[j] * inv;
        aff_out[((size_t)(b * 9 + j) << 16) + p] = a;
        if (vld[j]) {
            atomicAdd(&denl[j], a);
#pragma unroll
            for (int c = 0; c < FDIM_; ++c)
                atomicAdd(&numl[j][c], a * pv[c]);
        }
    }
    __syncthreads();

    if (t < 9 * FDIM_) {
        int j = t / FDIM_, c = t % FDIM_;
        int dr = j / 3 - 1, dc = j % 3 - 1;
        int rr = sy + dr, cc = sx + dc;
        if (((unsigned)rr < 16u) && ((unsigned)cc < 16u)) {
            int cid = rr * 16 + cc;
            atomicAdd(&num[((size_t)(b * NSPIX_) + cid) * FDIM_ + c], numl[j][c]);
        }
    }
    if (t < 9) {
        int dr = t / 3 - 1, dc = t % 3 - 1;
        int rr = sy + dr, cc = sx + dc;
        if (((unsigned)rr < 16u) && ((unsigned)cc < 16u))
            atomicAdd(&den[(size_t)(b * NSPIX_) + rr * 16 + cc], denl[t]);
    }
}

__global__ __launch_bounds__(256) void cent_update_kernel(const float* __restrict__ num,
                                                          const float* __restrict__ den,
                                                          float* __restrict__ cent) {
    int idx = blockIdx.x * 256 + threadIdx.x;      // B*256*20 = 10240
    if (idx >= B_ * NSPIX_ * FDIM_) return;
    int b = idx / (NSPIX_ * FDIM_);
    int r = idx - b * (NSPIX_ * FDIM_);
    int s = r / FDIM_, c = r % FDIM_;
    cent[((size_t)(b * FDIM_ + c) << 8) + s] =
        num[((size_t)(b * NSPIX_) + s) * FDIM_ + c] /
        (den[(size_t)(b * NSPIX_) + s] + 1e-16f);
}

// cid output: same candidate-id math as the reference, stored as float
__global__ __launch_bounds__(256) void cid_out_kernel(float* __restrict__ o) {
    int idx = blockIdx.x * 256 + threadIdx.x;      // 9*HW
    if (idx >= 9 * HW_) return;
    int j = idx >> 16, p = idx & 0xFFFF;
    int y = p >> 8, x = p & 255;
    int r0 = y >> 4, c0 = x >> 4;
    int dr = j / 3 - 1, dc = j % 3 - 1;
    int rr = r0 + dr; rr = rr < 0 ? 0 : (rr > 15 ? 15 : rr);
    int cc = c0 + dc; cc = cc < 0 ? 0 : (cc > 15 ? 15 : cc);
    o[idx] = (float)(rr * 16 + cc);
}

__global__ __launch_bounds__(256) void copy_kernel(const float* __restrict__ s,
                                                   float* __restrict__ d, size_t n) {
    size_t i = (size_t)blockIdx.x * 256 + threadIdx.x;
    if (i < n) d[i] = s[i];
}

// ---------------------------------------------------------------------------
// Host launcher
// ---------------------------------------------------------------------------
extern "C" void kernel_launch(void* const* d_in, const int* in_sizes, int n_in,
                              void* d_out, int out_size, void* d_ws, size_t ws_size,
                              hipStream_t stream) {
    (void)in_sizes; (void)n_in; (void)out_size; (void)ws_size;

    const float* x      = (const float*)d_in[0];
    const float* coords = (const float*)d_in[1];
    const float* feats  = (const float*)d_in[2];
    const float* W1     = (const float*)d_in[3];
    const float* g1     = (const float*)d_in[4];
    const float* b1     = (const float*)d_in[5];
    const float* W2     = (const float*)d_in[6];
    const float* g2     = (const float*)d_in[7];
    const float* b2     = (const float*)d_in[8];
    const float* W3     = (const float*)d_in[9];
    const float* g3     = (const float*)d_in[10];
    const float* b3     = (const float*)d_in[11];

    // workspace layout
    char* ws = (char*)d_ws;
    size_t off = 0;
    auto alloc = [&](size_t bytes) -> char* {
        char* r = ws + off;
        off += (bytes + 255) & ~(size_t)255;
        return r;
    };
    unsigned short* featsBF = (unsigned short*)alloc((size_t)B_ * CBB_ * HW_ * 2);
    unsigned short* w1nk    = (unsigned short*)alloc((size_t)KP1_ * NP1_ * 2);
    unsigned short* w2nk    = (unsigned short*)alloc((size_t)KP2_ * NP2_ * 2);
    unsigned short* w3nk    = (unsigned short*)alloc((size_t)KP3_ * NP3_ * 2);
    float* conv1o  = (float*)alloc((size_t)B_ * N1_ * HW_ * 4);
    unsigned short* h1bf = (unsigned short*)alloc((size_t)B_ * N1_ * HW_ * 2);
    float* conv2o  = (float*)alloc((size_t)B_ * N2_ * HW_ * 4);
    unsigned short* h2bf = (unsigned short*)alloc((size_t)B_ * N2_ * HW_ * 2);
    float* conv3o  = (float*)alloc((size_t)B_ * N3_ * HW_ * 4);
    float* pixelf  = (float*)alloc((size_t)B_ * FDIM_ * HW_ * 4);
    float* stats   = (float*)alloc(2 * 512 * 4);
    float* cent    = (float*)alloc((size_t)B_ * FDIM_ * NSPIX_ * 4);
    float* num     = (float*)alloc((size_t)B_ * NSPIX_ * FDIM_ * 4);
    float* den     = (float*)alloc((size_t)B_ * NSPIX_ * 4);

    float* out_aff  = (float*)d_out;                                  // B*9*HW
    float* out_cid  = out_aff + (size_t)B_ * 9 * HW_;                 // 9*HW
    float* out_cent = out_cid + (size_t)9 * HW_;                      // B*20*256

    // 1) feats -> bf16
    {
        size_t n = (size_t)B_ * CBB_ * HW_;
        cvt_bf16_kernel<<<(unsigned)((n + 255) / 256), 256, 0, stream>>>(feats, featsBF, n);
    }
    // 2) weight repacks (N-major [Npad][Kpad])
    {
        size_t n1 = (size_t)KP1_ * NP1_;
        repack_w_kernel<<<(unsigned)((n1 + 255) / 256), 256, 0, stream>>>(W1, w1nk, CBB_, N1_, K1_, KP1_, NP1_);
        size_t n2 = (size_t)KP2_ * NP2_;
        repack_w_kernel<<<(unsigned)((n2 + 255) / 256), 256, 0, stream>>>(W2, w2nk, N1_, N2_, K2_, KP2_, NP2_);
        size_t n3 = (size_t)KP3_ * NP3_;
        repack_w_kernel<<<(unsigned)((n3 + 255) / 256), 256, 0, stream>>>(W3, w3nk, N2_, N3_, K3_, KP3_, NP3_);
    }
    const unsigned mtiles = (B_ * HW_) / TM;   // 1024

    // 3) conv1 + BN + ReLU -> h1 bf16
    conv3x3_wmma_kernel<<<dim3(mtiles, NP1_ / TN), 256, 0, stream>>>(
        featsBF, w1nk, conv1o, CBB_, K1_, KP1_, NP1_, N1_);
    bn_stats_kernel<<<N1_, 256, 0, stream>>>(conv1o, stats, N1_);
    bn_relu_bf16_kernel<<<(unsigned)((size_t)B_ * N1_ * HW_ / 256), 256, 0, stream>>>(
        conv1o, stats, g1, b1, h1bf, N1_);

    // 4) conv2 + BN + ReLU -> h2 bf16
    conv3x3_wmma_kernel<<<dim3(mtiles, NP2_ / TN), 256, 0, stream>>>(
        h1bf, w2nk, conv2o, N1_, K2_, KP2_, NP2_, N2_);
    bn_stats_kernel<<<N2_, 256, 0, stream>>>(conv2o, stats, N2_);
    bn_relu_bf16_kernel<<<(unsigned)((size_t)B_ * N2_ * HW_ / 256), 256, 0, stream>>>(
        conv2o, stats, g2, b2, h2bf, N2_);

    // 5) conv3 + BN + ReLU -> pixel_f ch[0,15)
    conv3x3_wmma_kernel<<<dim3(mtiles, NP3_ / TN), 256, 0, stream>>>(
        h2bf, w3nk, conv3o, N2_, K3_, KP3_, NP3_, N3_);
    bn_stats_kernel<<<N3_, 256, 0, stream>>>(conv3o, stats, N3_);
    bn_relu_pf_kernel<<<(unsigned)((size_t)B_ * N3_ * HW_ / 256), 256, 0, stream>>>(
        conv3o, stats, g3, b3, pixelf);

    // 6) concat x + coords into pixel_f ch[15,20)
    concat_pf_kernel<<<(unsigned)((size_t)B_ * 5 * HW_ / 256), 256, 0, stream>>>(
        x, coords, pixelf);

    // 7) initial centroids
    cent_init_kernel<<<B_ * NSPIX_, 256, 0, stream>>>(pixelf, cent);

    // 8) SSN iterations
    for (int it = 0; it < NITER_; ++it) {
        size_t nn = (size_t)B_ * NSPIX_ * FDIM_;
        zero_kernel<<<(unsigned)((nn + 255) / 256), 256, 0, stream>>>(num, nn);
        size_t nd = (size_t)B_ * NSPIX_;
        zero_kernel<<<(unsigned)((nd + 255) / 256), 256, 0, stream>>>(den, nd);
        ssn_step_kernel<<<B_ * NSPIX_, 256, 0, stream>>>(pixelf, cent, num, den, out_aff);
        cent_update_kernel<<<(B_ * NSPIX_ * FDIM_ + 255) / 256, 256, 0, stream>>>(num, den, cent);
    }

    // 9) cid + cent outputs
    cid_out_kernel<<<(9 * HW_ + 255) / 256, 256, 0, stream>>>(out_cid);
    copy_kernel<<<(B_ * FDIM_ * NSPIX_ + 255) / 256, 256, 0, stream>>>(
        cent, out_cent, (size_t)B_ * FDIM_ * NSPIX_);
}
